// ScaledDotProductAttention_15573551415607
// MI455X (gfx1250) — matrix operand order repaired and checked by reading, not compiled, not run on hardware
//
#include <hip/hip_runtime.h>

// Fused scaled-dot-product attention for MI455X (gfx1250).
// B=32, S=2048, D=128, fp32 in/out. Outputs: O [B,S,D] then attn [B,S,S].
//  - HBM-bound on the 537MB write-once attn tensor -> non-temporal stores so
//    the K/V working set (64MB, all batches) stays resident in the 192MB L2.
//  - v_wmma_f32_16x16x32_f16 (f16 operands, f32 accumulate) for both matmuls.
//  - K/V chunks staged LDS-side via GLOBAL_LOAD_ASYNC_TO_LDS_B128 with
//    double-buffered prefetch hidden behind WMMA compute / softmax.

#define BATCH 32
#define SEQ   2048
#define DIM   128
#define TM    16      // query rows per workgroup (one WMMA M tile)
#define TCH   128     // K/V t-chunk staged in LDS (8 waves x 16 cols/tiles)
#define NCHUNK (SEQ / TCH)
#define G_ITERS 16    // per-thread async b128 issues per 64KB chunk

typedef __attribute__((ext_vector_type(16))) _Float16 v16h;
typedef __attribute__((ext_vector_type(8)))  _Float16 v8h;
typedef __attribute__((ext_vector_type(4)))  _Float16 v4h;
typedef __attribute__((ext_vector_type(8)))  float    v8f;
typedef __attribute__((ext_vector_type(4)))  float    v4f;

__device__ __forceinline__ v16h cat8(v8h lo, v8h hi) {
  v16h r;
#pragma unroll
  for (int i = 0; i < 8; ++i) { r[i] = lo[i]; r[i + 8] = hi[i]; }
  return r;
}

// Async-copy one 64KB chunk (TCH*DIM f32) global -> LDS.
// 256 threads x G_ITERS x 16B, GVS addressing (s[base] + v_off32).
// Tracked on ASYNCcnt; caller pairs with s_wait_asynccnt + barrier.
__device__ __forceinline__ void async_stage_chunk(const float* gbase,
                                                  const float* lbase, int tid) {
  const unsigned lds0 = (unsigned)(size_t)lbase + (unsigned)tid * 16u;
  const unsigned off0 = (unsigned)tid * 16u;
#pragma unroll
  for (int i = 0; i < G_ITERS; ++i) {
    asm volatile("global_load_async_to_lds_b128 %0, %1, %2"
                 :: "v"(lds0 + (unsigned)i * 4096u),
                    "v"(off0 + (unsigned)i * 4096u),
                    "s"(gbase)
                 : "memory");
  }
}

__device__ __forceinline__ void wait_async0() {
  asm volatile("s_wait_asynccnt 0x0" ::: "memory");
}

__global__ __launch_bounds__(256, 1)
void sdpa_fused_kernel(const float* __restrict__ Q,
                       const float* __restrict__ K,
                       const float* __restrict__ V,
                       float* __restrict__ outO,
                       float* __restrict__ outA)
{
  // LDS: 4KB Q(f16) + 128KB score strip(f32) + 2x64KB async K/V chunks + 1KB
  __shared__ __align__(16) _Float16 sQ[TM * DIM];
  __shared__ __align__(16) float    sS[TM * SEQ];
  __shared__ __align__(16) float    sKV[2][TCH * DIM];
  __shared__ __align__(16) float    sRed[TM * 16];

  const int tid  = threadIdx.x;
  const int lane = tid & 31;
  const int wave = tid >> 5;
  const int laneM  = lane & 15;   // M for A / N for B / column for C-D tiles
  const int hiHalf = lane >> 4;   // 0: lanes 0-15, 1: lanes 16-31

  const int mblk = blockIdx.x % (SEQ / TM);
  const int b    = blockIdx.x / (SEQ / TM);
  const int m0   = mblk * TM;
  const float inv_scale = 0.022097086912079608f; // 1/sqrt(2048) per reference

  const float* Kb = K + (size_t)b * SEQ * DIM;
  const float* Vb = V + (size_t)b * SEQ * DIM;

  // Kick off async copy of K chunk 0 immediately, stage Q meanwhile.
  async_stage_chunk(Kb, sKV[0], tid);

  const float* Qt = Q + ((size_t)b * SEQ + m0) * DIM;
  for (int i = tid; i < TM * DIM / 4; i += 256) {
    v4f q = *(const v4f*)(Qt + i * 4);
    *(v4h*)(sQ + i * 4) = __builtin_convertvector(q, v4h);
  }

  // ================= Phase 1: S = (Q K^T) * inv_scale =================
  for (int chunk = 0; chunk < NCHUNK; ++chunk) {
    wait_async0();          // my chunk-c async writes (issued last iter) done
    __syncthreads();        // everyone's done -> buffer readable, old reads done
    const float* sK = sKV[chunk & 1];
    if (chunk + 1 < NCHUNK) // prefetch next chunk while we compute
      async_stage_chunk(Kb + (size_t)(chunk + 1) * TCH * DIM,
                        sKV[(chunk + 1) & 1], tid);

    const int t0   = chunk * TCH;
    const int tloc = wave * 16;          // this wave's 16 t-columns
    v8f acc = {};
#pragma unroll
    for (int kk = 0; kk < DIM; kk += 32) {
      // A (Q 16x32 f16): lane holds M=laneM; elems 0-7 = K kb..kb+7, 8-15 = kb+16..+23
      const int kb = kk + hiHalf * 8;
      const _Float16* arow = sQ + laneM * DIM + kb;
      v16h a = cat8(*(const v8h*)(arow), *(const v8h*)(arow + 16));
      // B (K^T 32x16): lane holds N=t=tloc+laneM; elems 0-7 = K kb..kb+7, 8-15 = kb+16..+23 -> cvt
      const float* brow = sK + (tloc + laneM) * DIM + kk + hiHalf * 8;
      v4h h0 = __builtin_convertvector(*(const v4f*)(brow),      v4h);
      v4h h1 = __builtin_convertvector(*(const v4f*)(brow + 4),  v4h);
      v4h h2 = __builtin_convertvector(*(const v4f*)(brow + 16), v4h);
      v4h h3 = __builtin_convertvector(*(const v4f*)(brow + 20), v4h);
      v16h bm;
#pragma unroll
      for (int j = 0; j < 4; ++j) {
        bm[j] = h0[j]; bm[4 + j] = h1[j]; bm[8 + j] = h2[j]; bm[12 + j] = h3[j];
      }
      acc = __builtin_amdgcn_wmma_f32_16x16x32_f16(false, a, false, bm,
                                                   (short)0, acc, false, false);
    }
    // C/D layout: VGPR v -> row v (lanes 0-15) or v+8 (lanes 16-31), col laneM
#pragma unroll
    for (int v = 0; v < 8; ++v) {
      const int m = hiHalf ? (v + 8) : v;
      sS[m * SEQ + t0 + tloc + laneM] = acc[v] * inv_scale;
    }
  }
  __syncthreads();   // score strip complete

  // Overlap the V chunk-0 copy with the whole softmax phase.
  async_stage_chunk(Vb, sKV[0], tid);

  // ================= Softmax over each of the 16 rows (len 2048) =======
  {
    const int r = tid >> 4;              // row 0..15
    const int c = tid & 15;              // 16 threads/row, 128 cols each
    float* row = sS + r * SEQ + c * 128;

    float mx = -3.4e38f;
    for (int i = 0; i < 128; i += 4) {
      v4f s = *(const v4f*)(row + i);
      mx = fmaxf(mx, fmaxf(fmaxf(s[0], s[1]), fmaxf(s[2], s[3])));
    }
    sRed[tid] = mx;
    __syncthreads();
    mx = sRed[r * 16];
#pragma unroll
    for (int i = 1; i < 16; ++i) mx = fmaxf(mx, sRed[r * 16 + i]);
    __syncthreads();                     // protect sRed reuse (WAR)

    float sum = 0.f;
    for (int i = 0; i < 128; i += 4) {
      v4f s = *(const v4f*)(row + i);
      v4f e;
#pragma unroll
      for (int j = 0; j < 4; ++j) { e[j] = __expf(s[j] - mx); sum += e[j]; }
      *(v4f*)(row + i) = e;
    }
    sRed[tid] = sum;
    __syncthreads();
    sum = 0.f;
#pragma unroll
    for (int i = 0; i < 16; ++i) sum += sRed[r * 16 + i];
    const float inv = 1.0f / sum;

    // Normalize in LDS; stream attn to HBM non-temporally (write-once data,
    // must NOT evict the K/V working set from L2).
    float* arow = outA + ((size_t)b * SEQ + m0 + r) * SEQ + c * 128;
    for (int i = 0; i < 128; i += 4) {
      v4f p = *(const v4f*)(row + i);
#pragma unroll
      for (int j = 0; j < 4; ++j) p[j] *= inv;
      *(v4f*)(row + i) = p;
      __builtin_nontemporal_store(p, (v4f*)(arow + i));
    }
  }

  // ================= Phase 2: O = P V =================================
  v8f oacc = {};
  const int d0 = wave * 16;              // this wave's 16-wide D slice
  for (int chunk = 0; chunk < NCHUNK; ++chunk) {
    wait_async0();
    __syncthreads();
    const float* sV = sKV[chunk & 1];
    if (chunk + 1 < NCHUNK)
      async_stage_chunk(Vb + (size_t)(chunk + 1) * TCH * DIM,
                        sKV[(chunk + 1) & 1], tid);

    const int t0 = chunk * TCH;
#pragma unroll
    for (int tk = 0; tk < TCH; tk += 32) {
      // A (P 16x32): f32 probabilities in sS -> f16 fragment on the fly
      const int kb = t0 + tk + hiHalf * 8;
      const float* prow = sS + laneM * SEQ + kb;
      v4h p0 = __builtin_convertvector(*(const v4f*)(prow),      v4h);
      v4h p1 = __builtin_convertvector(*(const v4f*)(prow + 4),  v4h);
      v4h p2 = __builtin_convertvector(*(const v4f*)(prow + 16), v4h);
      v4h p3 = __builtin_convertvector(*(const v4f*)(prow + 20), v4h);
      v16h a;
#pragma unroll
      for (int j = 0; j < 4; ++j) {
        a[j] = p0[j]; a[4 + j] = p1[j]; a[8 + j] = p2[j]; a[12 + j] = p3[j];
      }
      // B (V 32x16): lane holds N=d=d0+laneM; K(=t) strided by DIM in row-major V
      const float* bcol = sV + (size_t)(tk + hiHalf * 8) * DIM + d0 + laneM;
      v16h bm;
#pragma unroll
      for (int e = 0; e < 16; ++e) bm[e] = (_Float16)bcol[(e + (e >= 8 ? 8 : 0)) * DIM];
      oacc = __builtin_amdgcn_wmma_f32_16x16x32_f16(false, a, false, bm,
                                                    (short)0, oacc, false, false);
    }
  }

  // store O tile non-temporally: out[b][m0+M][d0+N]
  float* orow = outO + ((size_t)b * SEQ + m0) * DIM + d0;
#pragma unroll
  for (int v = 0; v < 8; ++v) {
    const int m = hiHalf ? (v + 8) : v;
    __builtin_nontemporal_store(oacc[v], orow + (size_t)m * DIM + laneM);
  }
}

extern "C" void kernel_launch(void* const* d_in, const int* in_sizes, int n_in,
                              void* d_out, int out_size, void* d_ws, size_t ws_size,
                              hipStream_t stream) {
  (void)in_sizes; (void)n_in; (void)out_size; (void)d_ws; (void)ws_size;
  const float* Q = (const float*)d_in[0];
  const float* K = (const float*)d_in[1];
  const float* V = (const float*)d_in[2];
  float* outO = (float*)d_out;                               // [B,S,D]
  float* outA = outO + (size_t)BATCH * SEQ * DIM;            // [B,S,S]
  dim3 grid(BATCH * (SEQ / TM));
  sdpa_fused_kernel<<<grid, 256, 0, stream>>>(Q, K, V, outO, outA);
}
